// FinalLayer_19696720019880
// MI455X (gfx1250) — matrix operand order repaired and checked by reading, not compiled, not run on hardware
//
#include <hip/hip_runtime.h>

typedef __attribute__((ext_vector_type(16))) _Float16 v16h;
typedef __attribute__((ext_vector_type(8)))  float    v8f;

#define NB      4
#define NGRID   512
#define NTARGET 512
#define NBASIS  8
#define NCH     8

// One block = 256 threads = 8 waves. Each wave owns one 16-row t-tile of one
// (b,c) slice. Grid = (B*C) * 4 blocks = 128 blocks (each block covers 8 of
// the 32 t-tiles of its slice).
__global__ __launch_bounds__(256)
void rbf_setconv_head_kernel(const float* __restrict__ x_grid,    // (B,G,C)
                             const float* __restrict__ h_grid,    // (B,G,K,C)
                             const float* __restrict__ target_x,  // (B,T,C)
                             const float* __restrict__ W,         // (OUT=1,K)
                             const float* __restrict__ bias,      // (1,)
                             const float* __restrict__ sigma,     // (K,C)
                             float* __restrict__ out)             // (B,T,C)
{
    __shared__ float    s_x[NGRID];            // x_grid slice for (b,c)
    __shared__ _Float16 s_v[NGRID * NBASIS];   // V[(g,k)] = h*W, f16

    const int bc   = blockIdx.x >> 2;   // 0..31
    const int b    = bc >> 3;           // 0..3
    const int c    = bc & 7;            // 0..7
    const int tgrp = blockIdx.x & 3;    // which group of 8 t-tiles

    const int tid = threadIdx.x;

    // ---- cooperative LDS staging (full EXEC, uniform loops) ----
    for (int i = tid; i < NGRID; i += 256)
        s_x[i] = x_grid[(b * NGRID + i) * NCH + c];
    for (int i = tid; i < NGRID * NBASIS; i += 256) {
        const int g = i >> 3, k = i & 7;
        const float v = h_grid[((b * NGRID + g) * NBASIS + k) * NCH + c] * W[k];
        s_v[i] = (_Float16)v;
    }
    __syncthreads();

    // per-basis coefficient: -0.5 / (exp(sigma)+eps)^2  (natural-exp form)
    float coef[NBASIS];
#pragma unroll
    for (int k = 0; k < NBASIS; ++k) {
        const float s = __expf(sigma[k * NCH + c]) + 1e-6f;
        coef[k] = -0.5f / (s * s);
    }

    const int  lane   = tid & 31;
    const int  wv     = tid >> 5;
    const int  t0     = (tgrp * 8 + wv) * 16;  // base target row of this wave's tile
    const int  trow   = lane & 15;             // M row this lane contributes to A
    const bool hiHalf = (lane & 16) != 0;
    const float tx    = target_x[(b * NTARGET + t0 + trow) * NCH + c];

    v8f acc = {};
    // K' = (g,k) flattened, 4096 terms, 32 per WMMA -> 128 iterations
#pragma unroll 1
    for (int it = 0; it < (NGRID * NBASIS) / 32; ++it) {
        const int g0 = it * 4;
        float d2[4];
#pragma unroll
        for (int gg = 0; gg < 4; ++gg) {
            const float d = s_x[g0 + gg] - tx;
            d2[gg] = d * d;
        }

        v16h a, bb;
#pragma unroll
        for (int j = 0; j < 16; ++j) {
            // A fragment: 16-bit A 16x32 layout.
            //   lanes 0-15  cover local K' {0..7, 16..23}
            //   lanes 16-31 cover local K' {8..15, 24..31}
            const int qA = (j < 8 ? j : j + 8) + (hiHalf ? 8 : 0);
            const int gl = qA >> 3;        // local grid point 0..3
            const int k  = qA & 7;         // basis index
            a[j] = (_Float16)__expf(coef[k] * d2[gl]);

            // B fragment: 32x16 layout, lanes 0-15 hold K={0..7,16..23}, lanes 16-31
            // hold K={8..15,24..31}; V broadcast across all 16 columns.
            const int qB = (j < 8 ? j : j + 8) + (hiHalf ? 8 : 0);
            bb[j] = s_v[it * 32 + qB];
        }

        acc = __builtin_amdgcn_wmma_f32_16x16x32_f16(
            /*neg_a=*/false, a, /*neg_b=*/false, bb,
            /*c_mod=*/(short)0, acc, /*reuse_a=*/false, /*reuse_b=*/false);
    }

    // D layout (f32 16x16): lane n = column n; VGPR r -> row r (lanes 0-15)
    // or row r+8 (lanes 16-31). All columns identical (B was broadcast), so
    // lane 0 writes rows 0..7 and lane 16 writes rows 8..15.
    if (lane == 0 || lane == 16) {
        const float bs    = bias[0];
        const int   mbase = hiHalf ? 8 : 0;
#pragma unroll
        for (int r = 0; r < 8; ++r)
            out[(b * NTARGET + t0 + mbase + r) * NCH + c] = acc[r] + bs;
    }
}

extern "C" void kernel_launch(void* const* d_in, const int* in_sizes, int n_in,
                              void* d_out, int out_size, void* d_ws, size_t ws_size,
                              hipStream_t stream) {
    const float* x_grid   = (const float*)d_in[0];
    const float* h_grid   = (const float*)d_in[1];
    const float* target_x = (const float*)d_in[2];
    const float* W        = (const float*)d_in[3];
    const float* bias     = (const float*)d_in[4];
    const float* sigma    = (const float*)d_in[5];
    float*       out      = (float*)d_out;

    dim3 grid(128);   // (B*C)=32 slices * 4 blocks each
    dim3 block(256);  // 8 waves of 32
    rbf_setconv_head_kernel<<<grid, block, 0, stream>>>(
        x_grid, h_grid, target_x, W, bias, sigma, out);
}
